// DConv2d_68143951118429
// MI455X (gfx1250) — matrix operand order repaired and checked by reading, not compiled, not run on hardware
//
#include <hip/hip_runtime.h>

// Problem constants (match reference)
#define B_    4
#define C_IN  64
#define H_    128
#define W_    128
#define C_OUT 128
#define KK    9
#define HO    128
#define WO    128
#define HW    16384            // H*W == HO*WO
#define KDIM  576              // C_IN * KK (GEMM K dimension)
#define NT    32               // output positions per workgroup (N tile)
#define KSTR  592              // bf16 row stride in LDS (592*2 = 1184 B, 32B aligned)

typedef __attribute__((ext_vector_type(16))) __bf16 v16bf;
typedef __attribute__((ext_vector_type(8)))  float  v8f;

static __device__ __forceinline__ unsigned short f32_to_bf16(float f) {
  union { float f; unsigned int u; } v; v.f = f;
  unsigned int u = v.u;
  u += 0x7fffu + ((u >> 16) & 1u);   // round-to-nearest-even
  return (unsigned short)(u >> 16);
}

// ---------------------------------------------------------------------------
// Kernel 0: pack w_def (fp32 [C_OUT][KDIM]) into bf16 WMMA A-fragment order.
// Fragment v16bf index = (o_tile*18 + kb)*32 + lane ; within lane, slot j:
//   K = kb*32 + j + 16*(lane>=16)
// ---------------------------------------------------------------------------
__global__ __launch_bounds__(256) void pack_wdef_kernel(
    const float* __restrict__ w_def, unsigned short* __restrict__ wpack) {
  int idx = blockIdx.x * 256 + threadIdx.x;
  if (idx >= 8 * 18 * 32 * 16) return;
  int j    = idx & 15;
  int lane = (idx >> 4) & 31;
  int rest = idx >> 9;           // = o_tile*18 + kb
  int kb   = rest % 18;
  int ot   = rest / 18;
  int o    = ot * 16 + (lane & 15);
  int k    = kb * 32 + j + 16 * (lane >> 4);
  wpack[idx] = f32_to_bf16(w_def[o * KDIM + k]);
}

// ---------------------------------------------------------------------------
// Kernel 1: offset conv (3x3, pad 1, 18 out channels), fp32 direct conv.
// One thread per (b, ho, wo); w_off cached in LDS (41.5 KB).
// ---------------------------------------------------------------------------
__global__ __launch_bounds__(256) void offset_conv_kernel(
    const float* __restrict__ x, const float* __restrict__ w_off,
    const float* __restrict__ b_off, float* __restrict__ off) {
  __shared__ float wsh[18 * KDIM];
  int tid = threadIdx.x;
  for (int i = tid; i < 18 * KDIM; i += 256) wsh[i] = w_off[i];
  __syncthreads();

  int p  = blockIdx.x * 256 + tid;
  int b  = p / HW, q = p % HW;
  int ho = q / WO, wo = q % WO;

  float acc[18];
  #pragma unroll
  for (int oc = 0; oc < 18; ++oc) acc[oc] = b_off[oc];

  const float* xb = x + (size_t)b * C_IN * HW;
  for (int kh = 0; kh < 3; ++kh) {
    int y = ho - 1 + kh;
    if ((unsigned)y >= (unsigned)H_) continue;
    for (int kw = 0; kw < 3; ++kw) {
      int xc = wo - 1 + kw;
      if ((unsigned)xc >= (unsigned)W_) continue;
      int widx = kh * 3 + kw;
      for (int c = 0; c < C_IN; ++c) {
        float xv = xb[c * HW + y * W_ + xc];
        const float* wr = &wsh[c * KK + widx];
        #pragma unroll
        for (int oc = 0; oc < 18; ++oc) acc[oc] += xv * wr[oc * KDIM];
      }
    }
  }
  float* ob = off + (size_t)b * 18 * HW + q;
  #pragma unroll
  for (int oc = 0; oc < 18; ++oc) ob[oc * HW] = acc[oc];
}

// ---------------------------------------------------------------------------
// Kernel 2: bilinear sampling -> bf16 LDS tile -> WMMA GEMM -> out + bias.
// Workgroup = 256 threads (8 waves): NT=32 positions x C_OUT=128 channels.
// Wave w owns output channels [16w, 16w+16); 18 K-chunks x 2 N-subtiles.
// ---------------------------------------------------------------------------
__global__ __launch_bounds__(256) void dcn_wmma_kernel(
    const float* __restrict__ x, const float* __restrict__ off,
    const unsigned short* __restrict__ wpack, const float* __restrict__ b_def,
    float* __restrict__ out) {
  __shared__ unsigned short s[NT * KSTR];   // 37,888 bytes

  int tid = threadIdx.x;
  int p0  = blockIdx.x * NT;                // 32 consecutive flat positions
  int b   = p0 / HW;
  int q0  = p0 % HW;
  int ho  = q0 / WO;
  int wo0 = q0 % WO;                        // NT | WO, tile stays in one row
  const float* xb = x + (size_t)b * C_IN * HW;

  // ---- Stage 1: bilinear sampling into LDS as bf16, s[n][c*9+kpt] ----
  for (int task = tid; task < NT * KK; task += 256) {
    int n   = task / KK, kpt = task % KK;
    int q   = q0 + n;
    int wo  = wo0 + n;
    int kh  = kpt / 3, kw = kpt % 3;
    float dy = off[((size_t)b * 18 + 2 * kpt    ) * HW + q];
    float dx = off[((size_t)b * 18 + 2 * kpt + 1) * HW + q];
    float py = (float)(ho - 1 + kh) + dy;
    float px = (float)(wo - 1 + kw) + dx;
    float y0f = floorf(py), x0f = floorf(px);
    int   y0  = (int)y0f,   x0  = (int)x0f;
    int   y1  = y0 + 1,     x1  = x0 + 1;
    float wy1 = py - y0f, wx1 = px - x0f;
    float wy0 = 1.f - wy1, wx0 = 1.f - wx1;
    float vy0 = ((unsigned)y0 < (unsigned)H_) ? 1.f : 0.f;
    float vy1 = ((unsigned)y1 < (unsigned)H_) ? 1.f : 0.f;
    float vx0 = ((unsigned)x0 < (unsigned)W_) ? 1.f : 0.f;
    float vx1 = ((unsigned)x1 < (unsigned)W_) ? 1.f : 0.f;
    int y0c = y0 < 0 ? 0 : (y0 > H_ - 1 ? H_ - 1 : y0);
    int y1c = y1 < 0 ? 0 : (y1 > H_ - 1 ? H_ - 1 : y1);
    int x0c = x0 < 0 ? 0 : (x0 > W_ - 1 ? W_ - 1 : x0);
    int x1c = x1 < 0 ? 0 : (x1 > W_ - 1 ? W_ - 1 : x1);
    float w00 = wy0 * wx0 * vy0 * vx0, w01 = wy0 * wx1 * vy0 * vx1;
    float w10 = wy1 * wx0 * vy1 * vx0, w11 = wy1 * wx1 * vy1 * vx1;
    int i00 = y0c * W_ + x0c, i01 = y0c * W_ + x1c;
    int i10 = y1c * W_ + x0c, i11 = y1c * W_ + x1c;
    unsigned short* dst = &s[n * KSTR + kpt];
    for (int c = 0; c < C_IN; ++c) {
      const float* xc = xb + c * HW;
      float v = w00 * xc[i00] + w01 * xc[i01] + w10 * xc[i10] + w11 * xc[i11];
      dst[c * KK] = f32_to_bf16(v);
    }
  }
  __syncthreads();

  // ---- Stage 2: WMMA GEMM (bf16 in, f32 accumulate) ----
  int wave = tid >> 5, lane = tid & 31;
  int lhi  = lane >> 4, lo16 = lane & 15;
  v8f acc0 = {}; v8f acc1 = {};
  const v16bf* wp =
      reinterpret_cast<const v16bf*>(wpack) + (size_t)(wave * 18) * 32 + lane;
  for (int kb = 0; kb < 18; ++kb) {
    v16bf a  = wp[kb * 32];   // 32B contiguous global load, prepacked layout
    // B 32x16 bf16, contiguous K order matching the A pack: lane half selects K 0-15 / 16-31, N = lane%16
    v16bf b0 = *reinterpret_cast<const v16bf*>(
        &s[(lo16      ) * KSTR + kb * 32 + 16 * lhi]);
    v16bf b1 = *reinterpret_cast<const v16bf*>(
        &s[(16 + lo16 ) * KSTR + kb * 32 + 16 * lhi]);
    acc0 = __builtin_amdgcn_wmma_f32_16x16x32_bf16(
        false, a, false, b0, (short)0, acc0, false, false);
    acc1 = __builtin_amdgcn_wmma_f32_16x16x32_bf16(
        false, a, false, b1, (short)0, acc1, false, false);
  }

  // ---- Stage 3: epilogue (C layout: VGPR r -> M = r + 8*lhi, N = lo16) ----
  float bias[8];
  #pragma unroll
  for (int r = 0; r < 8; ++r) bias[r] = b_def[wave * 16 + r + 8 * lhi];
  #pragma unroll
  for (int sub = 0; sub < 2; ++sub) {
    v8f acc = sub ? acc1 : acc0;
    int q = q0 + sub * 16 + lo16;
    float* ob = out + (size_t)b * C_OUT * HW + q;
    #pragma unroll
    for (int r = 0; r < 8; ++r) {
      int o = wave * 16 + r + 8 * lhi;
      ob[(size_t)o * HW] = acc[r] + bias[r];
    }
  }
}

// ---------------------------------------------------------------------------
extern "C" void kernel_launch(void* const* d_in, const int* in_sizes, int n_in,
                              void* d_out, int out_size, void* d_ws, size_t ws_size,
                              hipStream_t stream) {
  (void)in_sizes; (void)n_in; (void)out_size; (void)ws_size;
  const float* x     = (const float*)d_in[0];
  const float* w_off = (const float*)d_in[1];
  const float* b_off = (const float*)d_in[2];
  const float* w_def = (const float*)d_in[3];
  const float* b_def = (const float*)d_in[4];
  float* out = (float*)d_out;

  // workspace layout: [0, 147456) packed bf16 weights; then offset map (fp32)
  unsigned short* wpack = (unsigned short*)d_ws;
  float* off = (float*)((char*)d_ws + (size_t)8 * 18 * 32 * 16 * 2);

  pack_wdef_kernel<<<(8 * 18 * 32 * 16 + 255) / 256, 256, 0, stream>>>(w_def, wpack);
  offset_conv_kernel<<<(B_ * HW) / 256, 256, 0, stream>>>(x, w_off, b_off, off);
  dcn_wmma_kernel<<<(B_ * HW) / NT, 256, 0, stream>>>(x, off, wpack, b_def, out);
}
